// G_Tensor3D_9474697855130
// MI455X (gfx1250) — matrix operand order repaired and checked by reading, not compiled, not run on hardware
//
#include <hip/hip_runtime.h>

typedef __attribute__((ext_vector_type(16))) _Float16 v16h;
typedef __attribute__((ext_vector_type(8)))  float    v8f;
typedef __attribute__((ext_vector_type(2)))  _Float16 h2v;

#define NPIX   (512*512)
#define BATCH  8

// ---------------------------------------------------------------------------
// Kernel 0: z_feat[b][k] = lerp of z_data rows (8x32 floats into workspace)
// ---------------------------------------------------------------------------
__global__ void zfeat_kernel(const float* __restrict__ z,
                             const float* __restrict__ z_data,
                             float* __restrict__ zf) {
  int t = threadIdx.x;            // 256 = 8*32 exactly
  int b = t >> 5, k = t & 31;
  float zn = 63.0f * z[b];
  int z0 = (int)zn;               // trunc toward zero, z in [0,1]
  z0 = max(0, min(z0, 63));
  int z1 = min(z0 + 1, 63);
  float zl = zn - truncf(zn);
  zf[t] = z_data[z0 * 32 + k] * (1.0f - zl) + z_data[z1 * 32 + k] * zl;
}

// ---------------------------------------------------------------------------
// Kernel 1: bilinear gather + 3-layer MLP via v_wmma_f32_16x16x32_f16.
// Block = 256 threads = 8 waves; wave w owns pixels [pix0+16w, pix0+16w+16).
// ---------------------------------------------------------------------------
__global__ __launch_bounds__(256)
void mlp_wmma_kernel(const float* __restrict__ data,
                     const float* __restrict__ lerp,
                     const int*   __restrict__ x0a, const int* __restrict__ y0a,
                     const int*   __restrict__ x1a, const int* __restrict__ y1a,
                     const float* __restrict__ W1,  const float* __restrict__ b1,
                     const float* __restrict__ W2,  const float* __restrict__ b2,
                     const float* __restrict__ W3,  const float* __restrict__ b3,
                     const float* __restrict__ zf,
                     float* __restrict__ out) {
  __shared__ float    s_xy[128 * 32];      // 16 KB: xy_feat for 128 pixels
  __shared__ float    s_z[BATCH * 32];     //  1 KB: z_feat
  __shared__ _Float16 s_h[8][16 * 32];     //  8 KB: per-wave activation staging

  const int tid  = threadIdx.x;
  const int lane = tid & 31;
  const int wave = tid >> 5;
  const int nlo  = lane & 15;              // column / row-in-tile index
  const int hi   = lane >> 4;              // lane half (0 or 1)
  const int pix0 = blockIdx.x * 128;

  s_z[tid] = zf[tid];

  // ---- cooperative bilinear gather: thread (lp,half) does 16 features -----
  {
    const int   lp   = tid >> 1;
    const int   half = tid & 1;
    const int   p    = pix0 + lp;
    const float wx = lerp[2 * p + 0];
    const float wy = lerp[2 * p + 1];
    const int ix0 = x0a[p], iy0 = y0a[p], ix1 = x1a[p], iy1 = y1a[p];
    const float w00 = (1.f - wx) * (1.f - wy), w01 = wx * (1.f - wy);
    const float w10 = (1.f - wx) * wy,         w11 = wx * wy;
    const float4* d00 = (const float4*)(data + ((size_t)iy0 * 512 + ix0) * 32 + half * 16);
    const float4* d01 = (const float4*)(data + ((size_t)iy0 * 512 + ix1) * 32 + half * 16);
    const float4* d10 = (const float4*)(data + ((size_t)iy1 * 512 + ix0) * 32 + half * 16);
    const float4* d11 = (const float4*)(data + ((size_t)iy1 * 512 + ix1) * 32 + half * 16);
    float4* dst = (float4*)&s_xy[lp * 32 + half * 16];
#pragma unroll
    for (int q = 0; q < 4; ++q) {
      float4 fa = d00[q], fb = d01[q], fc = d10[q], fd = d11[q];
      float4 r;
      r.x = fa.x * w00 + fb.x * w01 + fc.x * w10 + fd.x * w11;
      r.y = fa.y * w00 + fb.y * w01 + fc.y * w10 + fd.y * w11;
      r.z = fa.z * w00 + fb.z * w01 + fc.z * w10 + fd.z * w11;
      r.w = fa.w * w00 + fb.w * w01 + fc.w * w10 + fd.w * w11;
      dst[q] = r;
    }
  }

  // ---- loop-invariant weight B-fragments (ISA B layout: VGPR j = K rows
  // (2j | 2j+8 for j>=4) + 8*hi, pair across lanes(N)) ---------------------
  v16h B1a, B1b, B2a, B2b;
#pragma unroll
  for (int j = 0; j < 8; ++j) {
    const int kb = ((j < 4) ? 2 * j : 2 * j + 8) + hi * 8;
    B1a[2*j]   = (_Float16)W1[kb * 32 + nlo];
    B1a[2*j+1] = (_Float16)W1[(kb + 1) * 32 + nlo];
    B1b[2*j]   = (_Float16)W1[kb * 32 + nlo + 16];
    B1b[2*j+1] = (_Float16)W1[(kb + 1) * 32 + nlo + 16];
    B2a[2*j]   = (_Float16)W2[kb * 32 + nlo];
    B2a[2*j+1] = (_Float16)W2[(kb + 1) * 32 + nlo];
    B2b[2*j]   = (_Float16)W2[kb * 32 + nlo + 16];
    B2b[2*j+1] = (_Float16)W2[(kb + 1) * 32 + nlo + 16];
  }
  const float c1a = b1[nlo], c1b = b1[nlo + 16];
  const float c2a = b2[nlo], c2b = b2[nlo + 16];
  const float w3a = W3[nlo], w3b = W3[nlo + 16];
  const float c3  = b3[0];

  __syncthreads();

  const float*    xyrow = &s_xy[(wave * 16 + nlo) * 32];
  _Float16*       hrow  = &s_h[wave][0];

  for (int b = 0; b < BATCH; ++b) {
    const float* zrow = &s_z[b * 32];

    // A fragment (ISA 16-bit A layout): lane m holds row m; VGPR j holds
    // K pair (2j | 2j+8 for j>=4) + 8*hi.  feat = xy_feat * z_feat.
    v16h A;
#pragma unroll
    for (int j = 0; j < 8; ++j) {
      const int k0 = ((j < 4) ? 2 * j : 2 * j + 8) + hi * 8;
      A[2*j]   = (_Float16)(xyrow[k0]     * zrow[k0]);
      A[2*j+1] = (_Float16)(xyrow[k0 + 1] * zrow[k0 + 1]);
    }

    v8f C1a, C1b, C2a, C2b;
#pragma unroll
    for (int j = 0; j < 8; ++j) { C1a[j] = c1a; C1b[j] = c1b; C2a[j] = c2a; C2b[j] = c2b; }

    // ---- layer 1: h1 = relu(feat @ W1 + b1) : two 16x16x32 WMMAs ----------
    v8f D1a = __builtin_amdgcn_wmma_f32_16x16x32_f16(false, A, false, B1a, (short)0, C1a, false, false);
    v8f D1b = __builtin_amdgcn_wmma_f32_16x16x32_f16(false, A, false, B1b, (short)0, C1b, false, false);
#pragma unroll
    for (int j = 0; j < 8; ++j) { D1a[j] = fmaxf(D1a[j], 0.f); D1b[j] = fmaxf(D1b[j], 0.f); }

    // ---- D-layout -> A-layout transpose through LDS (f16, 1 KB/wave) ------
#pragma unroll
    for (int j = 0; j < 8; ++j) {
      const int M = hi * 8 + j;                      // D: lanes16-31 = rows 8..15
      hrow[M * 32 + nlo]      = (_Float16)D1a[j];
      hrow[M * 32 + nlo + 16] = (_Float16)D1b[j];
    }
    __syncthreads();
    v16h A2;
#pragma unroll
    for (int j = 0; j < 8; ++j) {
      const int k0 = ((j < 4) ? 2 * j : 2 * j + 8) + hi * 8;
      h2v pr = *(const h2v*)&hrow[nlo * 32 + k0];    // packed K pair, 4B aligned
      A2[2*j] = pr.x; A2[2*j+1] = pr.y;
    }
    __syncthreads();

    // ---- layer 2: h2 = relu(h1 @ W2 + b2) ---------------------------------
    v8f D2a = __builtin_amdgcn_wmma_f32_16x16x32_f16(false, A2, false, B2a, (short)0, C2a, false, false);
    v8f D2b = __builtin_amdgcn_wmma_f32_16x16x32_f16(false, A2, false, B2b, (short)0, C2b, false, false);
#pragma unroll
    for (int j = 0; j < 8; ++j) { D2a[j] = fmaxf(D2a[j], 0.f); D2b[j] = fmaxf(D2b[j], 0.f); }

    // ---- layer 3: out = h2 @ W3 + b3 : per-lane dot + 16-lane xor-reduce --
    float acc[8];
#pragma unroll
    for (int j = 0; j < 8; ++j) acc[j] = D2a[j] * w3a + D2b[j] * w3b;
#pragma unroll
    for (int m = 1; m < 16; m <<= 1) {
#pragma unroll
      for (int j = 0; j < 8; ++j) acc[j] += __shfl_xor(acc[j], m, 32);
    }
    if (nlo == 0) {                                  // lane0 -> rows 0..7, lane16 -> rows 8..15
      float* op = out + (size_t)b * NPIX + pix0 + wave * 16 + hi * 8;
      float4 lo4 = make_float4(acc[0] + c3, acc[1] + c3, acc[2] + c3, acc[3] + c3);
      float4 hi4 = make_float4(acc[4] + c3, acc[5] + c3, acc[6] + c3, acc[7] + c3);
      *(float4*)(op)     = lo4;
      *(float4*)(op + 4) = hi4;
    }
  }
}

// ---------------------------------------------------------------------------
extern "C" void kernel_launch(void* const* d_in, const int* in_sizes, int n_in,
                              void* d_out, int out_size, void* d_ws, size_t ws_size,
                              hipStream_t stream) {
  const float* z    = (const float*)d_in[0];
  const float* data = (const float*)d_in[1];
  const float* zd   = (const float*)d_in[2];
  const float* lw   = (const float*)d_in[3];
  const int*   x0   = (const int*)d_in[4];
  const int*   y0   = (const int*)d_in[5];
  const int*   x1   = (const int*)d_in[6];
  const int*   y1   = (const int*)d_in[7];
  const float* W1   = (const float*)d_in[8];
  const float* b1   = (const float*)d_in[9];
  const float* W2   = (const float*)d_in[10];
  const float* b2   = (const float*)d_in[11];
  const float* W3   = (const float*)d_in[12];
  const float* b3   = (const float*)d_in[13];
  float* out = (float*)d_out;
  float* zf  = (float*)d_ws;   // 8*32 floats = 1 KB

  zfeat_kernel<<<1, 256, 0, stream>>>(z, zd, zf);
  mlp_wmma_kernel<<<NPIX / 128, 256, 0, stream>>>(data, lw, x0, y0, x1, y1,
                                                  W1, b1, W2, b2, W3, b3,
                                                  zf, out);
}
